// GraphFactorization_66769561584074
// MI455X (gfx1250) — matrix operand order repaired and checked by reading, not compiled, not run on hardware
//
#include <hip/hip_runtime.h>

#define NN 8192
#define DD 128
#define LMBD_ 0.1f

typedef __attribute__((ext_vector_type(16))) _Float16 v16h;
typedef __attribute__((ext_vector_type(8)))  _Float16 v8h;
typedef __attribute__((ext_vector_type(8)))  float    v8f;

// ---------------------------------------------------------------------------
// Kernel 1: Y = f16(W + b); fp32 reg-loss partials per block.
// ---------------------------------------------------------------------------
__global__ __launch_bounds__(256) void gf_prep(const float* __restrict__ W,
                                               const float* __restrict__ b,
                                               _Float16* __restrict__ Y,
                                               float* __restrict__ regPartials) {
    __shared__ float red[256];
    const int idx = blockIdx.x * 256 + threadIdx.x;   // over NN*DD
    const float y = W[idx] + b[idx & (DD - 1)];
    Y[idx] = (_Float16)y;
    red[threadIdx.x] = 0.5f * LMBD_ * y * y;
    __syncthreads();
    for (int s = 128; s > 0; s >>= 1) {
        if (threadIdx.x < s) red[threadIdx.x] += red[threadIdx.x + s];
        __syncthreads();
    }
    if (threadIdx.x == 0) regPartials[blockIdx.x] = red[0];
}

// ---------------------------------------------------------------------------
// Kernel 2: masked residual over a dense WMMA Y·Yᵀ.
// Block tile 32(M) x 64(N); 8 waves, each computing one 16x16 tile via
// 4x v_wmma_f32_16x16x32_f16 (K = 128). A is streamed non-temporally.
// ---------------------------------------------------------------------------
__global__ __launch_bounds__(256) void gf_main(const float* __restrict__ A,
                                               const _Float16* __restrict__ Y,
                                               float* __restrict__ mainPartials) {
    const int wave = threadIdx.x >> 5;
    const int lane = threadIdx.x & 31;
    const int half = lane >> 4;      // lane group 0-15 / 16-31
    const int l16  = lane & 15;
    const int wm = wave >> 2;        // 0..1
    const int wn = wave & 3;         // 0..3
    const int m0 = blockIdx.y * 32 + wm * 16;
    const int n0 = blockIdx.x * 64 + wn * 16;

    // A-frag row for this lane: M = m0 + l16 ; B-frag (Yᵀ column) : N = n0 + l16
    const _Float16* rowA = Y + (size_t)(m0 + l16) * DD;
    const _Float16* rowB = Y + (size_t)(n0 + l16) * DD;

    v8f c = {};
#pragma unroll
    for (int k = 0; k < DD; k += 32) {
        // 16-bit A-matrix 16x32 layout: VGPR0-3 = K[k+8*half .. +7],
        //                               VGPR4-7 = K[k+16+8*half .. +7]
        union { v16h v; v8h h[2]; } af;
        af.h[0] = *(const v8h*)(rowA + k + 8 * half);
        af.h[1] = *(const v8h*)(rowA + k + 16 + 8 * half);
        // 16-bit B-matrix 32x16 layout: same K order as A (K[k+8*half..+7], K[k+16+8*half..+7]) of col l16
        union { v16h v; v8h h[2]; } bu;
        bu.h[0] = *(const v8h*)(rowB + k + 8 * half);
        bu.h[1] = *(const v8h*)(rowB + k + 16 + 8 * half);
        v16h bf = bu.v;
        c = __builtin_amdgcn_wmma_f32_16x16x32_f16(false, af.v, false, bf,
                                                   (short)0, c, false, false);
    }

    // C/D layout: VGPR r <-> row m0 + r + 8*half, lane <-> col n0 + l16
    float local = 0.f;
    const size_t base = (size_t)(m0 + 8 * half) * NN + (size_t)(n0 + l16);
#pragma unroll
    for (int r = 0; r < 8; ++r) {
        const float a = __builtin_nontemporal_load(A + base + (size_t)r * NN);
        const float d = a - c[r];
        local += (a > 0.f) ? 0.5f * d * d : 0.f;
    }

    // wave32 reduction
    for (int off = 16; off > 0; off >>= 1)
        local += __shfl_xor(local, off, 32);

    __shared__ float wred[8];
    if (lane == 0) wred[wave] = local;
    __syncthreads();
    if (threadIdx.x == 0) {
        float s = 0.f;
#pragma unroll
        for (int i = 0; i < 8; ++i) s += wred[i];
        mainPartials[blockIdx.y * gridDim.x + blockIdx.x] = s;
    }
}

// ---------------------------------------------------------------------------
// Kernel 3: final reduction of main + reg partials -> scalar loss.
// ---------------------------------------------------------------------------
__global__ __launch_bounds__(256) void gf_reduce(const float* __restrict__ mainPartials,
                                                 int nMain,
                                                 const float* __restrict__ regPartials,
                                                 int nReg,
                                                 float* __restrict__ out) {
    __shared__ float red[256];
    float s = 0.f;
    for (int i = threadIdx.x; i < nMain; i += 256) s += mainPartials[i];
    for (int i = threadIdx.x; i < nReg;  i += 256) s += regPartials[i];
    red[threadIdx.x] = s;
    __syncthreads();
    for (int st = 128; st > 0; st >>= 1) {
        if (threadIdx.x < st) red[threadIdx.x] += red[threadIdx.x + st];
        __syncthreads();
    }
    if (threadIdx.x == 0) out[0] = red[0];
}

// ---------------------------------------------------------------------------
extern "C" void kernel_launch(void* const* d_in, const int* in_sizes, int n_in,
                              void* d_out, int out_size, void* d_ws, size_t ws_size,
                              hipStream_t stream) {
    const float* A = (const float*)d_in[0];   // [NN, NN]
    const float* W = (const float*)d_in[1];   // [NN, DD]
    const float* b = (const float*)d_in[2];   // [DD]
    float* out = (float*)d_out;               // scalar

    // workspace layout: f16 Y (2 MB) | main partials (32768 f32) | reg partials (4096 f32)
    _Float16* Y = (_Float16*)d_ws;
    float* mainPartials = (float*)((char*)d_ws + (size_t)NN * DD * sizeof(_Float16));
    const int nMain = (NN / 64) * (NN / 32);  // 32768 blocks
    float* regPartials = mainPartials + nMain;
    const int nReg = (NN * DD) / 256;         // 4096 blocks

    gf_prep<<<nReg, 256, 0, stream>>>(W, b, Y, regPartials);
    gf_main<<<dim3(NN / 64, NN / 32), 256, 0, stream>>>(A, Y, mainPartials);
    gf_reduce<<<1, 256, 0, stream>>>(mainPartials, nMain, regPartials, nReg, out);
}
